// LigColumnAttention_73254962201331
// MI455X (gfx1250) — matrix operand order repaired and checked by reading, not compiled, not run on hardware
//
#include <hip/hip_runtime.h>
#include <hip/hip_bf16.h>

using bf16 = __bf16;
typedef __attribute__((ext_vector_type(16))) __bf16 v16bf;
typedef __attribute__((ext_vector_type(8)))  __bf16 v8bf;
typedef __attribute__((ext_vector_type(8)))  float  v8f;
typedef __attribute__((ext_vector_type(4)))  unsigned int v4u;
typedef __attribute__((ext_vector_type(8)))  int v8i;
typedef __attribute__((ext_vector_type(4)))  int v4i;

// Problem constants (B=1)
#define MDIM 256
#define IDIM 384
#define CIN  256
#define CATT 32
#define HH   8
#define NROWS (MDIM * IDIM)   // 98304

// ---------------------------------------------------------------------------
// WMMA fragment loaders (wave32; A and B share contiguous k = 16*(lane/16) + s)
// ---------------------------------------------------------------------------
__device__ __forceinline__ v16bf load_a_frag(const bf16* base, int ld, int m0, int k0) {
    const int lane = threadIdx.x & 31;
    const int row  = m0 + (lane & 15);
    const int off  = (lane >> 4) ? 16 : 0;
    const bf16* p  = base + (size_t)row * ld + k0 + off;
    union { v8bf h[2]; v16bf v; } u;
    u.h[0] = *(const v8bf*)(p);
    u.h[1] = *(const v8bf*)(p + 8);
    return u.v;
}

__device__ __forceinline__ v16bf load_b_frag(const bf16* baseT, int ld, int n0, int k0) {
    const int lane = threadIdx.x & 31;
    const int row  = n0 + (lane & 15);
    const int koff = k0 + ((lane >> 4) ? 16 : 0);
    return *(const v16bf*)(baseT + (size_t)row * ld + koff);
}

__device__ __forceinline__ v8f wmma_bf16(v16bf a, v16bf b, v8f c) {
    return __builtin_amdgcn_wmma_f32_16x16x32_bf16(false, a, false, b, (short)0, c, false, false);
}

// ---------------------------------------------------------------------------
// Tensor Data Mover: 1D contiguous global -> LDS copy of nbytes (bf16 data).
// D# per CDNA5 ISA: group0 = {count=1, lds_addr,
// global_addr, type=2}; group1 = {wg_mask=0, data_size=1(2B), tensor_dim0 =
// tile_dim0 = nelem, dim0_stride = nelem}. Groups 2/3 unused (zero).
// Wave-level op: call from one wave; tracked by TENSORcnt.
// ---------------------------------------------------------------------------
__device__ __forceinline__ void tdm_load_1d(unsigned lds_off, const void* gaddr,
                                            unsigned nbytes) {
    unsigned long long ga = (unsigned long long)gaddr;
    const unsigned nelem = nbytes >> 1;            // bf16 elements
    v4u g0;
    g0[0] = 1u;                                    // count=1, user descriptor
    g0[1] = lds_off;                               // LDS byte address
    g0[2] = (unsigned)(ga & 0xffffffffu);          // global_addr[31:0]
    g0[3] = (unsigned)((ga >> 32) & 0x01ffffffu) | (2u << 30);  // [56:32] | type=2
    v8i g1;
    g1[0] = (int)(1u << 16);                       // wg_mask=0, data_size=1 (2B)
    g1[1] = (int)((nelem & 0xffffu) << 16);        // abar=0 | tensor_dim0[15:0]
    g1[2] = (int)((nelem >> 16) & 0xffffu);        // tensor_dim0[31:16] | tdim1 lo=0
    g1[3] = (int)((nelem & 0xffffu) << 16);        // tdim1 hi=0 | tile_dim0
    g1[4] = 0;                                     // tile_dim1=0, tile_dim2=0
    g1[5] = (int)nelem;                            // tensor_dim0_stride[31:0]
    g1[6] = 0;
    g1[7] = 0;
    v4i z = {0, 0, 0, 0};
#if defined(__clang_major__) && (__clang_major__ >= 23)
    v8i z8 = {0, 0, 0, 0, 0, 0, 0, 0};
    __builtin_amdgcn_tensor_load_to_lds(g0, g1, z, z, z8, 0);
#else
    __builtin_amdgcn_tensor_load_to_lds(g0, g1, z, z, 0);
#endif
}

// ---------------------------------------------------------------------------
// Kernel 0: weight transpose + f32 -> bf16  (w: K x N -> wT: N x K)
// ---------------------------------------------------------------------------
__global__ __launch_bounds__(256) void k_transpose_bf16(const float* __restrict__ w,
                                                        bf16* __restrict__ wT,
                                                        int K, int N) {
    int idx = blockIdx.x * 256 + threadIdx.x;
    if (idx >= K * N) return;
    int k = idx / N, n = idx % N;
    wT[(size_t)n * K + k] = (bf16)w[idx];
}

// ---------------------------------------------------------------------------
// Kernel 1: LayerNorm over last dim (256) -> bf16. One wave per row.
// ---------------------------------------------------------------------------
__global__ __launch_bounds__(256) void k_layernorm(const float* __restrict__ x,
                                                   const float* __restrict__ g,
                                                   const float* __restrict__ b,
                                                   bf16* __restrict__ y) {
    const int row  = blockIdx.x * 8 + (threadIdx.x >> 5);
    const int lane = threadIdx.x & 31;
    const float* xr = x + (size_t)row * CIN;
    float4 v0 = *(const float4*)(xr + lane * 8);
    float4 v1 = *(const float4*)(xr + lane * 8 + 4);
    float va[8] = {v0.x, v0.y, v0.z, v0.w, v1.x, v1.y, v1.z, v1.w};
    float s = 0.f, s2 = 0.f;
    #pragma unroll
    for (int j = 0; j < 8; ++j) { s += va[j]; s2 += va[j] * va[j]; }
    #pragma unroll
    for (int m = 16; m >= 1; m >>= 1) {
        s  += __shfl_xor(s,  m, 32);
        s2 += __shfl_xor(s2, m, 32);
    }
    const float mu  = s * (1.0f / CIN);
    const float var = s2 * (1.0f / CIN) - mu * mu;
    const float inv = rsqrtf(var + 1e-5f);
    v8bf o;
    #pragma unroll
    for (int j = 0; j < 8; ++j) {
        int c = lane * 8 + j;
        o[j] = (bf16)((va[j] - mu) * inv * g[c] + b[c]);
    }
    *(v8bf*)(y + (size_t)row * CIN + lane * 8) = o;
}

// ---------------------------------------------------------------------------
// Generic WMMA GEMM: C = A[M x K] * BT[N x K]^T  (+epilogue)
// Block = 256 threads = 8 waves; WG tile 128(M) x 64(N); wave tile 16 x 64.
// EPI 1: +bias, sigmoid -> bf16       (gate)
// EPI 2: +bias -> f32                 (final output)
// EPI 3: scatter q/k/v into attention-native layouts:
//        q,k: [(i*8+h)][m][c]   v: [(i*8+h)][c][m]   (col = c*24 + part*8 + h)
// ---------------------------------------------------------------------------
template <int EPI>
__global__ __launch_bounds__(256) void k_gemm_wmma(const bf16* __restrict__ A,
                                                   const bf16* __restrict__ BT,
                                                   void* __restrict__ Cv,
                                                   bf16* __restrict__ P1,
                                                   bf16* __restrict__ P2,
                                                   const float* __restrict__ bias,
                                                   int K, int N) {
    const int lane = threadIdx.x & 31;
    const int wv   = threadIdx.x >> 5;
    const int m0   = blockIdx.x * 128 + wv * 16;
    const int n0   = blockIdx.y * 64;

    v8f acc[4] = {};
    for (int k0 = 0; k0 < K; k0 += 32) {
        if (k0 + 32 < K)
            __builtin_prefetch(A + (size_t)(m0 + (lane & 15)) * K + k0 + 32, 0, 0);
        v16bf a = load_a_frag(A, K, m0, k0);
        #pragma unroll
        for (int j = 0; j < 4; ++j) {
            v16bf bb = load_b_frag(BT, K, n0 + 16 * j, k0);
            acc[j] = wmma_bf16(a, bb, acc[j]);
        }
    }

    const int colc = lane & 15;
    const int rb   = (lane >> 4) * 8;
    #pragma unroll
    for (int j = 0; j < 4; ++j) {
        const int col = n0 + 16 * j + colc;
        const float bv = (EPI == 1 || EPI == 2) ? bias[col] : 0.f;
        #pragma unroll
        for (int r = 0; r < 8; ++r) {
            const int row = m0 + rb + r;
            float v = acc[j][r] + bv;
            if (EPI == 3) {
                const int m = row / IDIM, i = row - m * IDIM;
                const int c = col / 24, t = col - c * 24;
                const int part = t >> 3, h = t & 7;
                const size_t blk = (size_t)(i * HH + h);
                const bf16 bvv = (bf16)v;
                if (part == 0)      ((bf16*)Cv)[(blk * MDIM + m) * CATT + c] = bvv;
                else if (part == 1) P1[(blk * MDIM + m) * CATT + c] = bvv;
                else                P2[(blk * CATT + c) * MDIM + m] = bvv;
            } else {
                const size_t idx = (size_t)row * N + col;
                if (EPI == 1) ((bf16*)Cv)[idx] = (bf16)(1.f / (1.f + __expf(-v)));
                if (EPI == 2) ((float*)Cv)[idx] = v;
            }
        }
    }
}

// ---------------------------------------------------------------------------
// Kernel 3: fused attention per (i,h). One WG (8 waves), full 256x256
// attention: S = Q K^T / sqrt(32), row-softmax, O = P V, O *= gate.
// Q/K/V tiles arrive via three 16KB TDM (tensor_load_to_lds) transfers.
// LDS: Q(16K) K(16K) VT(16K) + per-wave f32 score strips (256K) + invsum (1K)
//  = 305 KB  (CDNA5 320KB WGP LDS).
// ---------------------------------------------------------------------------
#define SQ_OFF  0u
#define SK_OFF  16384u
#define SVT_OFF 32768u
#define SS_OFF  49152u
#define ATTN_SMEM_BYTES (3 * 256 * 32 * 2 + 8 * 32 * 256 * 4 + 256 * 4)

__global__ __launch_bounds__(256) void k_attention(const bf16* __restrict__ qbuf,
                                                   const bf16* __restrict__ kbuf,
                                                   const bf16* __restrict__ vbuf,
                                                   const bf16* __restrict__ gate,
                                                   bf16* __restrict__ attO) {
    extern __shared__ char smem[];
    bf16*  sQ   = (bf16*)(smem + SQ_OFF);        // 256 x 32 row-major
    bf16*  sK   = (bf16*)(smem + SK_OFF);        // 256 x 32 (BT for QK^T)
    bf16*  sVT  = (bf16*)(smem + SVT_OFF);       // 32 x 256 (BT for PV)
    float* sS   = (float*)(smem + SS_OFF);       // 8 x (32 x 256) f32 strips
    float* sInv = (float*)(smem + SS_OFF + 8 * 32 * 256 * 4);

    const int ih   = blockIdx.x;                 // i*8 + h
    const int i    = ih >> 3;
    const int h    = ih & 7;
    const int tid  = threadIdx.x;
    const int lane = tid & 31;
    const int w    = tid >> 5;

    // ---- stage Q, K, V^T: three contiguous 16KB TDM DMAs issued by wave 0
    if (w == 0) {
        const size_t blk = (size_t)ih * (MDIM * CATT);     // 8192 elements
        tdm_load_1d(SQ_OFF,  qbuf + blk, MDIM * CATT * 2);
        tdm_load_1d(SK_OFF,  kbuf + blk, MDIM * CATT * 2);
        tdm_load_1d(SVT_OFF, vbuf + blk, MDIM * CATT * 2);
        __builtin_amdgcn_s_wait_tensorcnt(0);
    }
    __syncthreads();

    // ---- S = Q K^T * 1/sqrt(32) for this wave's 32-row strip
    const float scale = 0.17677669529663689f;
    float* Sw = sS + (size_t)w * 32 * 256;
    const int colc = lane & 15;
    const int rb   = (lane >> 4) * 8;
    {
        v16bf a0 = load_a_frag(sQ, 32, 32 * w, 0);
        v16bf a1 = load_a_frag(sQ, 32, 32 * w + 16, 0);
        for (int nt = 0; nt < 16; ++nt) {
            const int n0 = nt * 16;
            v16bf bb = load_b_frag(sK, 32, n0, 0);
            v8f c0 = {}; v8f c1 = {};
            c0 = wmma_bf16(a0, bb, c0);
            c1 = wmma_bf16(a1, bb, c1);
            #pragma unroll
            for (int r = 0; r < 8; ++r) {
                Sw[(rb + r) * 256 + n0 + colc]      = c0[r] * scale;
                Sw[(16 + rb + r) * 256 + n0 + colc] = c1[r] * scale;
            }
        }
    }
    asm volatile("s_wait_dscnt 0" ::: "memory");   // in-wave LDS RAW fence

    // ---- row softmax: lane owns row `lane` of the strip; write unnormalized
    // exp() as bf16 in place (row stride stays 1024B), keep 1/sum in sInv.
    {
        float* srow = Sw + lane * 256;
        bf16*  prow = (bf16*)srow;
        float mx = -3.4e38f;
        for (int n = 0; n < 256; ++n) mx = fmaxf(mx, srow[n]);
        float sum = 0.f;
        for (int n0 = 0; n0 < 256; n0 += 32) {
            float t[32];
            #pragma unroll
            for (int j = 0; j < 32; ++j) t[j] = __expf(srow[n0 + j] - mx);
            asm volatile("" ::: "memory");         // order reads before aliased writes
            #pragma unroll
            for (int j = 0; j < 32; ++j) { sum += t[j]; prow[n0 + j] = (bf16)t[j]; }
        }
        sInv[w * 32 + lane] = 1.0f / sum;
    }
    asm volatile("s_wait_dscnt 0" ::: "memory");

    // ---- O = P V  (K = 256 over n); P row stride = 512 bf16 elements
    const bf16* P = (const bf16*)Sw;
    v8f o00 = {}, o01 = {}, o10 = {}, o11 = {};
    for (int k0 = 0; k0 < 256; k0 += 32) {
        v16bf a0 = load_a_frag(P, 512, 0,  k0);
        v16bf a1 = load_a_frag(P, 512, 16, k0);
        v16bf b0 = load_b_frag(sVT, 256, 0,  k0);
        v16bf b1 = load_b_frag(sVT, 256, 16, k0);
        o00 = wmma_bf16(a0, b0, o00);
        o01 = wmma_bf16(a0, b1, o01);
        o10 = wmma_bf16(a1, b0, o10);
        o11 = wmma_bf16(a1, b1, o11);
    }

    // ---- epilogue: normalize, gate, scatter to att layout col = c*8 + h
    v8f* oc[4] = {&o00, &o01, &o10, &o11};
    const int mt[4] = {0, 0, 16, 16};
    const int ct[4] = {0, 16, 0, 16};
    #pragma unroll
    for (int t = 0; t < 4; ++t) {
        #pragma unroll
        for (int r = 0; r < 8; ++r) {
            const int ml = mt[t] + rb + r;
            const int mg = 32 * w + ml;
            const int cg = ct[t] + colc;
            const size_t idx = ((size_t)mg * IDIM + i) * 256 + cg * 8 + h;
            const float val = (*oc[t])[r] * sInv[w * 32 + ml] * (float)gate[idx];
            attO[idx] = (bf16)val;
        }
    }
}

// ---------------------------------------------------------------------------
extern "C" void kernel_launch(void* const* d_in, const int* in_sizes, int n_in,
                              void* d_out, int out_size, void* d_ws, size_t ws_size,
                              hipStream_t stream) {
    const float* x1d     = (const float*)d_in[0];
    const float* ln_g    = (const float*)d_in[1];
    const float* ln_b    = (const float*)d_in[2];
    const float* w_qkv   = (const float*)d_in[3];
    const float* w_gate  = (const float*)d_in[4];
    const float* b_gate  = (const float*)d_in[5];
    const float* w_final = (const float*)d_in[6];
    const float* b_final = (const float*)d_in[7];

    char* ws = (char*)d_ws;
    bf16* x_ln  = (bf16*)ws; ws += (size_t)NROWS * CIN * 2;    // 50.3 MB
    bf16* wqkvT = (bf16*)ws; ws += (size_t)768 * 256 * 2;
    bf16* wgT   = (bf16*)ws; ws += (size_t)256 * 256 * 2;
    bf16* wfT   = (bf16*)ws; ws += (size_t)256 * 256 * 2;
    bf16* qbuf  = (bf16*)ws; ws += (size_t)NROWS * 256 * 2;    // [i*8+h][m][c]
    bf16* kbuf  = (bf16*)ws; ws += (size_t)NROWS * 256 * 2;
    bf16* vbuf  = (bf16*)ws; ws += (size_t)NROWS * 256 * 2;    // [i*8+h][c][m]
    bf16* gateb = (bf16*)ws; ws += (size_t)NROWS * 256 * 2;
    bf16* attO  = (bf16*)ws;

    // weight prep: transpose to N x K bf16 (B-fragment friendly)
    k_transpose_bf16<<<(256 * 768 + 255) / 256, 256, 0, stream>>>(w_qkv, wqkvT, 256, 768);
    k_transpose_bf16<<<(256 * 256 + 255) / 256, 256, 0, stream>>>(w_gate, wgT, 256, 256);
    k_transpose_bf16<<<(256 * 256 + 255) / 256, 256, 0, stream>>>(w_final, wfT, 256, 256);

    // LayerNorm -> bf16
    k_layernorm<<<NROWS / 8, 256, 0, stream>>>(x1d, ln_g, ln_b, x_ln);

    // QKV GEMM with attention-layout scatter epilogue: 98304 x 768
    {
        dim3 g(NROWS / 128, 768 / 64);
        k_gemm_wmma<3><<<g, 256, 0, stream>>>(x_ln, wqkvT, (void*)qbuf, kbuf, vbuf,
                                              nullptr, 256, 768);
    }
    // Gate GEMM + sigmoid: 98304 x 256
    {
        dim3 g(NROWS / 128, 256 / 64);
        k_gemm_wmma<1><<<g, 256, 0, stream>>>(x_ln, wgT, (void*)gateb, nullptr, nullptr,
                                              b_gate, 256, 256);
    }
    // Fused attention: one WG per (i, h); Q/K/V staged via TDM
    k_attention<<<IDIM * HH, 256, ATTN_SMEM_BYTES, stream>>>(qbuf, kbuf, vbuf, gateb, attO);

    // Final GEMM + bias -> f32 output
    {
        dim3 g(NROWS / 128, 256 / 64);
        k_gemm_wmma<2><<<g, 256, 0, stream>>>(attO, wfT, d_out, nullptr, nullptr,
                                              b_final, 256, 256);
    }
    (void)in_sizes; (void)n_in; (void)out_size; (void)ws_size;
}
